// ResidualHiddenAdapter_47167330844868
// MI455X (gfx1250) — matrix operand order repaired and checked by reading, not compiled, not run on hardware
//
#include <hip/hip_runtime.h>
#include <hip/hip_bf16.h>

typedef __attribute__((ext_vector_type(16))) __bf16 v16bf;
typedef __attribute__((ext_vector_type(8)))  __bf16 v8bf;
typedef __attribute__((ext_vector_type(8)))  float  v8f;
typedef __attribute__((ext_vector_type(4)))  float  v4f;
typedef __attribute__((ext_vector_type(4)))  int    v4i;

#define NROWS 16384
#define VSZ   256
#define FEAT  2048
#define HID   4096
#define OUTD  2048
#define LMAX  56
#define NEG_INF_F (-1e30f)

// GEMM tiling: block 128x256, 8 waves of 64x64, K-step 64, double-buffered LDS.
#define GBM 128
#define GBN 256
#define GBK 64

union BFrag { v16bf v; v8bf h[2]; };

// ---------------------------------------------------------------------------
// gfx1250 async global->LDS copy (ASYNCcnt path) with portable fallback.
// Builtin signature (from compiler diagnostic): param0 = AS1 int4* (global src),
// param1 = AS3 int4* (LDS dst), then two int immediates (offset, cpol).
// ---------------------------------------------------------------------------
#if defined(__has_builtin)
#if __has_builtin(__builtin_amdgcn_global_load_async_to_lds_b128)
#define USE_ASYNC_COPY 1
#endif
#endif
#ifndef USE_ASYNC_COPY
#define USE_ASYNC_COPY 0
#endif

typedef __attribute__((address_space(1))) v4i* global_v4i_ptr;
typedef __attribute__((address_space(3))) v4i* lds_v4i_ptr;

__device__ __forceinline__ void cp_b128(const __bf16* g, __bf16* l) {
#if USE_ASYNC_COPY
    __builtin_amdgcn_global_load_async_to_lds_b128(
        (global_v4i_ptr)(unsigned long long)(size_t)g,
        (lds_v4i_ptr)(unsigned int)(size_t)l,
        0, 0);
#else
    *reinterpret_cast<v8bf*>(l) = *reinterpret_cast<const v8bf*>(g);
#endif
}

__device__ __forceinline__ void wait_cp_async() {
#if USE_ASYNC_COPY
#if __has_builtin(__builtin_amdgcn_s_wait_asynccnt)
    __builtin_amdgcn_s_wait_asynccnt(0);
#else
    asm volatile("s_wait_asynccnt 0x0" ::: "memory");
#endif
#endif
}

// ---------------------------------------------------------------------------
// Pre-conversion kernels (run once per launch; negligible vs GEMM time).
// ---------------------------------------------------------------------------
__global__ __launch_bounds__(256)
void cvt_bf16_kernel(const float* __restrict__ src, __bf16* __restrict__ dst,
                     long long n)
{
    long long i = ((long long)blockIdx.x * 256 + threadIdx.x) * 8;
    if (i + 8 > n) return;
    v4f a = *reinterpret_cast<const v4f*>(src + i);
    v4f b = *reinterpret_cast<const v4f*>(src + i + 4);
    v8bf o;
    o[0] = (__bf16)a[0]; o[1] = (__bf16)a[1]; o[2] = (__bf16)a[2]; o[3] = (__bf16)a[3];
    o[4] = (__bf16)b[0]; o[5] = (__bf16)b[1]; o[6] = (__bf16)b[2]; o[7] = (__bf16)b[3];
    *reinterpret_cast<v8bf*>(dst + i) = o;
}

// dst[N,K] = bf16(src[K,N]) -- 64x64 LDS tile transpose, coalesced both sides.
__global__ __launch_bounds__(256)
void transpose_cvt_kernel(const float* __restrict__ src, __bf16* __restrict__ dst,
                          int K, int N)
{
    __shared__ __align__(16) __bf16 Ts[64][66];
    const int n0 = blockIdx.x * 64;
    const int k0 = blockIdx.y * 64;
    const int tid = threadIdx.x;
    {
        const int rk = tid >> 2;              // 0..63
        const int cn = (tid & 3) * 16;        // 0..48
        const float* g = src + (size_t)(k0 + rk) * N + n0 + cn;
        #pragma unroll
        for (int j = 0; j < 4; ++j) {
            v4f f = *reinterpret_cast<const v4f*>(g + j * 4);
            Ts[rk][cn + j * 4 + 0] = (__bf16)f[0];
            Ts[rk][cn + j * 4 + 1] = (__bf16)f[1];
            Ts[rk][cn + j * 4 + 2] = (__bf16)f[2];
            Ts[rk][cn + j * 4 + 3] = (__bf16)f[3];
        }
    }
    __syncthreads();
    {
        const int rn = tid >> 2;
        const int ck = (tid & 3) * 16;
        __bf16* o = dst + (size_t)(n0 + rn) * K + k0 + ck;
        v8bf w0, w1;
        #pragma unroll
        for (int j = 0; j < 8; ++j) w0[j] = Ts[ck + j][rn];
        #pragma unroll
        for (int j = 0; j < 8; ++j) w1[j] = Ts[ck + 8 + j][rn];
        *reinterpret_cast<v8bf*>(o)     = w0;
        *reinterpret_cast<v8bf*>(o + 8) = w1;
    }
}

// ---------------------------------------------------------------------------
// Confidence features: wave32 per row, masked softmax / top-2 / entropy.
// ---------------------------------------------------------------------------
__global__ __launch_bounds__(256)
void conf_feats_kernel(const float* __restrict__ logits,
                       const int* __restrict__ dom,
                       const int* __restrict__ cols,
                       const unsigned char* __restrict__ mask,
                       float* __restrict__ cfe)
{
    const int wid  = threadIdx.x >> 5;
    const int lane = threadIdx.x & 31;
    const int i = blockIdx.x * 8 + wid;

    const int d = dom[i];
    const float* lrow = logits + (size_t)i * VSZ;

    const int l0 = lane, l1 = lane + 32;
    const bool v0 = (l0 < LMAX) && (mask[d * LMAX + l0] != 0);
    const bool v1 = (l1 < LMAX) && (mask[d * LMAX + l1] != 0);
    float x0 = v0 ? lrow[cols[d * LMAX + l0]] : NEG_INF_F;
    float x1 = v1 ? lrow[cols[d * LMAX + l1]] : NEG_INF_F;

    float m = fmaxf(x0, x1);
    #pragma unroll
    for (int s = 16; s >= 1; s >>= 1) m = fmaxf(m, __shfl_xor(m, s, 32));

    float e0 = v0 ? __expf(x0 - m) : 0.0f;
    float e1 = v1 ? __expf(x1 - m) : 0.0f;
    float ssum = e0 + e1;
    #pragma unroll
    for (int s = 16; s >= 1; s >>= 1) ssum += __shfl_xor(ssum, s, 32);

    const float inv  = 1.0f / ssum;
    const float logs = __logf(ssum);
    const float p0 = e0 * inv, p1 = e1 * inv;

    float ent = 0.0f;
    if (v0) ent -= p0 * (x0 - m - logs);
    if (v1) ent -= p1 * (x1 - m - logs);
    #pragma unroll
    for (int s = 16; s >= 1; s >>= 1) ent += __shfl_xor(ent, s, 32);

    float a = fmaxf(p0, p1), b = fminf(p0, p1);
    #pragma unroll
    for (int s = 16; s >= 1; s >>= 1) {
        float oa = __shfl_xor(a, s, 32);
        float ob = __shfl_xor(b, s, 32);
        float nb = fmaxf(fminf(a, oa), fmaxf(b, ob));
        a = fmaxf(a, oa); b = nb;
    }

    if (lane == 0) {
        cfe[i * 4 + 0] = a;
        cfe[i * 4 + 1] = a - b;
        cfe[i * 4 + 2] = ent;
        cfe[i * 4 + 3] = 0.0f;
    }
}

// ---------------------------------------------------------------------------
// Shared GEMM mainloop pieces (A: [M,K] bf16 row-major, B: [N,K] bf16 row-major)
// ---------------------------------------------------------------------------
#define DECLARE_TILES() \
    __shared__ __align__(16) __bf16 As[2][GBM][GBK]; \
    __shared__ __align__(16) __bf16 Bs[2][GBN][GBK]

#define STAGE_TILE(buf, kk)                                                     \
    do {                                                                        \
        const __bf16* ga_ = Abase + (size_t)(m0 + arow) * lda + (kk) + asub;    \
        __bf16* la_ = &As[buf][arow][asub];                                     \
        _Pragma("unroll")                                                       \
        for (int j = 0; j < 4; ++j) cp_b128(ga_ + j * 8, la_ + j * 8);          \
        const __bf16* gb_ = Bbase + (size_t)(n0 + tid) * ldb + (kk);            \
        __bf16* lb_ = &Bs[buf][tid][0];                                         \
        _Pragma("unroll")                                                       \
        for (int j = 0; j < 8; ++j) cp_b128(gb_ + j * 8, lb_ + j * 8);          \
    } while (0)

#define COMPUTE_TILE(buf)                                                       \
    do {                                                                        \
        _Pragma("unroll")                                                       \
        for (int s = 0; s < 2; ++s) {                                           \
            BFrag af[4], bfm[4];                                                \
            _Pragma("unroll")                                                   \
            for (int fm = 0; fm < 4; ++fm) {                                    \
                const __bf16* p = &As[buf][wm + fm * 16 + mrow][s * 32];        \
                af[fm].h[0] = *reinterpret_cast<const v8bf*>(p + kbA);          \
                af[fm].h[1] = *reinterpret_cast<const v8bf*>(p + kbA + 16);     \
            }                                                                   \
            _Pragma("unroll")                                                   \
            for (int fn = 0; fn < 4; ++fn) {                                    \
                const __bf16* p = &Bs[buf][wn + fn * 16 + mrow][s * 32];        \
                bfm[fn].h[0] = *reinterpret_cast<const v8bf*>(p + kbB);         \
                bfm[fn].h[1] = *reinterpret_cast<const v8bf*>(p + kbB + 16);    \
            }                                                                   \
            _Pragma("unroll")                                                   \
            for (int fm = 0; fm < 4; ++fm)                                      \
                _Pragma("unroll")                                               \
                for (int fn = 0; fn < 4; ++fn)                                  \
                    acc[fm][fn] = __builtin_amdgcn_wmma_f32_16x16x32_bf16(      \
                        false, af[fm].v, false, bfm[fn].v,                      \
                        (short)0, acc[fm][fn], false, false);                   \
        }                                                                       \
    } while (0)

#define GEMM_PREAMBLE()                                                         \
    const int tid  = threadIdx.x;                                               \
    const int wid  = tid >> 5;                                                  \
    const int lane = tid & 31;                                                  \
    const int wm = (wid & 1) * 64;                                              \
    const int wn = (wid >> 1) * 64;                                             \
    const int m0 = blockIdx.x * GBM;                                            \
    const int n0 = blockIdx.y * GBN;                                            \
    const int arow = tid >> 1;                                                  \
    const int asub = (tid & 1) * 32;                                            \
    const int mrow = lane & 15;                                                 \
    const int kbA  = (lane >> 4) * 8;                                           \
    const int kbB  = (lane >> 4) * 8;                                           \
    v8f acc[4][4] = {}

#define GEMM_MAINLOOP(KDIM)                                                     \
    do {                                                                        \
        const int NT = (KDIM) / GBK;                                            \
        STAGE_TILE(0, 0);                                                       \
        for (int kt = 0; kt < NT; ++kt) {                                       \
            const int cur = kt & 1;                                             \
            wait_cp_async();                                                    \
            __syncthreads();                                                    \
            if (kt + 1 < NT) STAGE_TILE(cur ^ 1, (kt + 1) * GBK);               \
            COMPUTE_TILE(cur);                                                  \
            __syncthreads();                                                    \
        }                                                                       \
    } while (0)

// ---------------------------------------------------------------------------
// GEMM1: h = relu( feats @ W1[0:2048] + c @ W1[2048:2051] + b1 )  -> bf16
// ---------------------------------------------------------------------------
__global__ __launch_bounds__(256)
void gemm1_relu_kernel(const __bf16* __restrict__ Abase,   // feats bf16 [M,FEAT]
                       const __bf16* __restrict__ Bbase,   // W1^T bf16 [HID,FEAT]
                       const float* __restrict__ W1,       // original (for last 3 rows)
                       const float* __restrict__ b1,
                       const float* __restrict__ cfe,
                       __bf16* __restrict__ hbuf)
{
    DECLARE_TILES();
    GEMM_PREAMBLE();
    const int lda = FEAT, ldb = FEAT;
    GEMM_MAINLOOP(FEAT);

    const int rbase = (lane >> 4) * 8;
    const int cl    = lane & 15;
    const float* W1e = W1 + (size_t)FEAT * HID;
    #pragma unroll
    for (int fm = 0; fm < 4; ++fm) {
        #pragma unroll
        for (int r = 0; r < 8; ++r) {
            const int row = m0 + wm + fm * 16 + rbase + r;
            v4f c = *reinterpret_cast<const v4f*>(cfe + row * 4);
            #pragma unroll
            for (int fn = 0; fn < 4; ++fn) {
                const int col = n0 + wn + fn * 16 + cl;
                float v = acc[fm][fn][r];
                v += c[0] * W1e[col]
                   + c[1] * W1e[HID + col]
                   + c[2] * W1e[2 * HID + col]
                   + b1[col];
                v = fmaxf(v, 0.0f);
                hbuf[(size_t)row * HID + col] = (__bf16)v;
            }
        }
    }
}

// ---------------------------------------------------------------------------
// GEMM2: out = h_base + alpha * (h @ W2 + b2)
// ---------------------------------------------------------------------------
__global__ __launch_bounds__(256)
void gemm2_residual_kernel(const __bf16* __restrict__ Abase,  // h bf16 [M,HID]
                           const __bf16* __restrict__ Bbase,  // W2^T bf16 [OUTD,HID]
                           const float* __restrict__ b2,
                           const float* __restrict__ h_base,
                           const float* __restrict__ alpha_p,
                           float* __restrict__ out)
{
    DECLARE_TILES();
    GEMM_PREAMBLE();
    const int lda = HID, ldb = HID;
    const float alpha = alpha_p[0];
    GEMM_MAINLOOP(HID);

    const int rbase = (lane >> 4) * 8;
    const int cl    = lane & 15;
    #pragma unroll
    for (int fm = 0; fm < 4; ++fm) {
        #pragma unroll
        for (int r = 0; r < 8; ++r) {
            const int row = m0 + wm + fm * 16 + rbase + r;
            #pragma unroll
            for (int fn = 0; fn < 4; ++fn) {
                const int col = n0 + wn + fn * 16 + cl;
                const size_t idx = (size_t)row * OUTD + col;
                out[idx] = h_base[idx] + alpha * (acc[fm][fn][r] + b2[col]);
            }
        }
    }
}

// ---------------------------------------------------------------------------
extern "C" void kernel_launch(void* const* d_in, const int* in_sizes, int n_in,
                              void* d_out, int out_size, void* d_ws, size_t ws_size,
                              hipStream_t stream) {
    const float* h_base        = (const float*)d_in[0];
    const int*   domain_ids    = (const int*)d_in[1];
    const float* global_logits = (const float*)d_in[2];
    const float* feats         = (const float*)d_in[3];
    const int*   cols_table    = (const int*)d_in[4];
    const unsigned char* col_mask = (const unsigned char*)d_in[5];
    const float* W1            = (const float*)d_in[6];
    const float* b1            = (const float*)d_in[7];
    const float* W2            = (const float*)d_in[8];
    const float* b2            = (const float*)d_in[9];
    const float* alpha         = (const float*)d_in[10];
    float* out = (float*)d_out;

    char* ws = (char*)d_ws;
    size_t off = 0;
    float*  cfe     = (float*)(ws + off);  off += (size_t)NROWS * 4 * 4;
    __bf16* hbuf    = (__bf16*)(ws + off); off += (size_t)NROWS * HID * 2;
    __bf16* featsbf = (__bf16*)(ws + off); off += (size_t)NROWS * FEAT * 2;
    __bf16* w1t     = (__bf16*)(ws + off); off += (size_t)HID * FEAT * 2;
    __bf16* w2t     = (__bf16*)(ws + off); off += (size_t)OUTD * HID * 2;

    // One-time precision conversion / weight transposes (bf16, [N,K] layout).
    const long long nfeat = (long long)NROWS * FEAT;
    cvt_bf16_kernel<<<(unsigned)(nfeat / 2048), 256, 0, stream>>>(feats, featsbf, nfeat);
    transpose_cvt_kernel<<<dim3(HID / 64, FEAT / 64), 256, 0, stream>>>(W1, w1t, FEAT, HID);
    transpose_cvt_kernel<<<dim3(OUTD / 64, HID / 64), 256, 0, stream>>>(W2, w2t, HID, OUTD);

    conf_feats_kernel<<<NROWS / 8, 256, 0, stream>>>(
        global_logits, domain_ids, cols_table, col_mask, cfe);

    gemm1_relu_kernel<<<dim3(NROWS / GBM, HID / GBN), 256, 0, stream>>>(
        featsbf, w1t, W1, b1, cfe, hbuf);

    gemm2_residual_kernel<<<dim3(NROWS / GBM, OUTD / GBN), 256, 0, stream>>>(
        hbuf, w2t, b2, h_base, alpha, out);
}
